// MogLSTM_49289044689353
// MI455X (gfx1250) — compile-verified, matrix operand order checked by reading, not run on hardware
//
#include <hip/hip_runtime.h>
#include <math.h>

typedef __attribute__((ext_vector_type(16))) _Float16 v16h;
typedef __attribute__((ext_vector_type(8)))  float    v8f;

// Problem constants (match the reference).
constexpr int Bc = 512, Sc = 256, INc = 300, IN2c = 300, Hc = 512, FNc = 3;

// GEMM tiling: 128x128 block tile, K-step 32, 256 threads = 8 wave32 waves,
// wave grid 4(M) x 2(N): each wave computes 32x64 = 2x4 fragments of 16x16.
#define BM 128
#define BN 128
#define BK 32
#define NTHREADS 256

enum { EPI_TANH16 = 0, EPI_MOGX = 1, EPI_MOGH = 2, EPI_GATES1 = 3, EPI_GATES2 = 4 };

struct EpiArgs {
    const float*    b1;        // bias 1 (per output column)
    const float*    b2;        // bias 2 (per output column)
    const _Float16* aux;       // f16 elementwise multiplicand (mogrifier)
    int             aux_ld;
    _Float16*       out16;     // f16 output
    int             out16_ld;
    float*          outf;      // f32 output (gates)
    int             outf_ld;
};

__device__ __forceinline__ float sigf(float x) {
    return 1.0f / (1.0f + __expf(-x));
}

template <typename TA, typename TB, int EPI>
__global__ __launch_bounds__(NTHREADS) void gemm_wmma(
    const TA* __restrict__ A, int lda,
    const TB* __restrict__ B, int ldb,
    int M, int N, int K, EpiArgs ea)
{
    __shared__ _Float16 As[BM * BK];   // [row][k]
    __shared__ _Float16 Bs[BK * BN];   // [k][col]

    const int tid  = threadIdx.x;
    const int lane = tid & 31;
    const int wid  = tid >> 5;
    const int wm   = (wid & 3) * 32;   // wave M offset inside tile
    const int wn   = (wid >> 2) * 64;  // wave N offset inside tile
    const int m0   = blockIdx.y * BM;
    const int n0   = blockIdx.x * BN;

    const int rl = lane & 15;          // row/col within 16 group
    const int kh = (lane >> 4) << 3;   // 0 or 8 : K-half / M-half selector

    v8f acc[2][4];
    {
        v8f z = {};
        #pragma unroll
        for (int i = 0; i < 2; ++i)
            #pragma unroll
            for (int j = 0; j < 4; ++j) acc[i][j] = z;
    }

    for (int k0 = 0; k0 < K; k0 += BK) {
        // ---- stage A tile (BM x BK), convert to f16, zero-pad OOB ----
        #pragma unroll
        for (int i = 0; i < (BM * BK) / NTHREADS; ++i) {
            int e = tid + NTHREADS * i;
            int r = e >> 5, c = e & 31;
            int gm = m0 + r, gk = k0 + c;
            float val = (gm < M && gk < K) ? (float)A[(size_t)gm * lda + gk] : 0.0f;
            As[r * BK + c] = (_Float16)val;
        }
        // ---- stage B tile (BK x BN) ----
        #pragma unroll
        for (int i = 0; i < (BK * BN) / NTHREADS; ++i) {
            int e = tid + NTHREADS * i;
            int r = e >> 7, c = e & 127;
            int gk = k0 + r, gn = n0 + c;
            float val = (gk < K && gn < N) ? (float)B[(size_t)gk * ldb + gn] : 0.0f;
            Bs[r * BN + c] = (_Float16)val;
        }
        __syncthreads();

        // ---- build fragments per ISA 16-bit A/B layouts ----
        v16h af[2], bf[4];
        #pragma unroll
        for (int mi = 0; mi < 2; ++mi) {
            const _Float16* p = &As[(wm + mi * 16 + rl) * BK];
            #pragma unroll
            for (int q = 0; q < 8; ++q) {
                af[mi][q]     = p[kh + q];           // K = kh .. kh+7
                af[mi][8 + q] = p[16 + kh + q];      // K = 16+kh .. 16+kh+7
            }
        }
        #pragma unroll
        for (int ni = 0; ni < 4; ++ni) {
            const _Float16* p = &Bs[wn + ni * 16 + rl];
            #pragma unroll
            for (int q = 0; q < 8; ++q) {
                bf[ni][q]     = p[(kh + q) * BN];
                bf[ni][8 + q] = p[(16 + kh + q) * BN];
            }
        }

        #pragma unroll
        for (int mi = 0; mi < 2; ++mi)
            #pragma unroll
            for (int ni = 0; ni < 4; ++ni)
                acc[mi][ni] = __builtin_amdgcn_wmma_f32_16x16x32_f16(
                    false, af[mi], false, bf[ni], (short)0, acc[mi][ni], false, false);

        __syncthreads();
    }

    // ---- fused epilogue; C/D layout: VGPR v -> row (kh + v), lane&15 -> col ----
    #pragma unroll
    for (int mi = 0; mi < 2; ++mi) {
        #pragma unroll
        for (int ni = 0; ni < 4; ++ni) {
            int gn = n0 + wn + ni * 16 + rl;
            #pragma unroll
            for (int vv = 0; vv < 8; ++vv) {
                int gm = m0 + wm + mi * 16 + kh + vv;
                if (gm < M && gn < N) {
                    float a = acc[mi][ni][vv];
                    if constexpr (EPI == EPI_TANH16) {
                        ea.out16[(size_t)gm * ea.out16_ld + gn] =
                            (_Float16)tanhf(a + ea.b1[gn]);
                    } else if constexpr (EPI == EPI_MOGX || EPI == EPI_MOGH) {
                        float s = 2.0f * sigf(a);
                        float x = (float)ea.aux[(size_t)gm * ea.aux_ld + gn];
                        ea.out16[(size_t)gm * ea.out16_ld + gn] = (_Float16)(s * x);
                    } else if constexpr (EPI == EPI_GATES1) {
                        ea.outf[(size_t)gm * ea.outf_ld + gn] = a + ea.b1[gn] + ea.b2[gn];
                    } else { // EPI_GATES2: accumulate
                        ea.outf[(size_t)gm * ea.outf_ld + gn] += a;
                    }
                }
            }
        }
    }
}

// ---- init: h=0, c=0, convmax=-inf ----
__global__ void init_state(_Float16* h16, float* c32, float* convmax, int nh, int ncm)
{
    int i = blockIdx.x * blockDim.x + threadIdx.x;
    if (i < nh) { h16[i] = (_Float16)0.0f; c32[i] = 0.0f; }
    if (i < ncm) convmax[i] = -1e30f;
}

// ---- LSTM cell pointwise update ----
__global__ __launch_bounds__(256) void lstm_cell(
    const float* __restrict__ gates, float* __restrict__ c32,
    _Float16* __restrict__ h16, float* __restrict__ hring, int t)
{
    int i = blockIdx.x * blockDim.x + threadIdx.x;
    if (i >= Bc * Hc) return;
    int b = i >> 9;          // / Hc
    int j = i & (Hc - 1);    // % Hc
    const float* g = gates + (size_t)b * (4 * Hc);
    float ig = sigf(g[j]);
    float fg = sigf(g[Hc + j]);
    float gg = tanhf(g[2 * Hc + j]);
    float og = sigf(g[3 * Hc + j]);
    float cn = fg * c32[i] + ig * gg;
    c32[i] = cn;
    float hn = og * tanhf(cn);
    h16[i] = (_Float16)hn;
    hring[(size_t)(t % 5) * (Bc * Hc) + i] = hn;
}

// ---- conv windows ending at step t: relu + running global max ----
__global__ __launch_bounds__(256) void conv_update(
    const float* __restrict__ hring,
    const float* __restrict__ Wc3, const float* __restrict__ bc3,
    const float* __restrict__ Wc4, const float* __restrict__ bc4,
    const float* __restrict__ Wc5, const float* __restrict__ bc5,
    float* __restrict__ convmax, int t)
{
    __shared__ float red[256];
    const int b   = blockIdx.x;
    const int tid = threadIdx.x;
    const float* wsets[3] = {Wc3, Wc4, Wc5};
    const float* bsets[3] = {bc3, bc4, bc5};

    for (int fi = 0; fi < 3; ++fi) {
        const int fsz = 3 + fi;
        const int p = t - (fsz - 1);       // window start position
        if (p < 0) continue;
        for (int fn = 0; fn < FNc; ++fn) {
            float part = 0.0f;
            for (int k = 0; k < fsz; ++k) {
                const float* hrow = hring + (size_t)((p + k) % 5) * (Bc * Hc) + (size_t)b * Hc;
                const float* wrow = wsets[fi] + ((size_t)fn * fsz + k) * Hc;
                for (int h = tid; h < Hc; h += 256) part += hrow[h] * wrow[h];
            }
            red[tid] = part;
            __syncthreads();
            for (int s = 128; s > 0; s >>= 1) {
                if (tid < s) red[tid] += red[tid + s];
                __syncthreads();
            }
            if (tid == 0) {
                float y = red[0] + bsets[fi][fn];
                y = y > 0.0f ? y : 0.0f;
                int oi = b * (3 * FNc) + fi * FNc + fn;
                if (y > convmax[oi]) convmax[oi] = y;
            }
            __syncthreads();
        }
    }
}

// ---- final 9 -> 2 linear ----
__global__ void final_linear(const float* __restrict__ convmax,
                             const float* __restrict__ Wl, const float* __restrict__ bl,
                             float* __restrict__ out)
{
    int i = blockIdx.x * blockDim.x + threadIdx.x;
    if (i >= Bc * 2) return;
    int b = i >> 1, c = i & 1;
    float a = bl[c];
    #pragma unroll
    for (int j = 0; j < 3 * FNc; ++j) a += convmax[b * (3 * FNc) + j] * Wl[j * 2 + c];
    out[i] = a;
}

extern "C" void kernel_launch(void* const* d_in, const int* in_sizes, int n_in,
                              void* d_out, int out_size, void* d_ws, size_t ws_size,
                              hipStream_t stream)
{
    (void)in_sizes; (void)n_in; (void)out_size; (void)ws_size;

    const float* x    = (const float*)d_in[0];
    const float* W_in = (const float*)d_in[1];
    const float* b_in = (const float*)d_in[2];
    const float* Wih  = (const float*)d_in[3];
    const float* Whh  = (const float*)d_in[4];
    const float* bih  = (const float*)d_in[5];
    const float* bhh  = (const float*)d_in[6];
    const float* Q    = (const float*)d_in[7];
    const float* R    = (const float*)d_in[8];
    const float* Wc3  = (const float*)d_in[9];
    const float* bc3  = (const float*)d_in[10];
    const float* Wc4  = (const float*)d_in[11];
    const float* bc4  = (const float*)d_in[12];
    const float* Wc5  = (const float*)d_in[13];
    const float* bc5  = (const float*)d_in[14];
    const float* Wlin = (const float*)d_in[15];
    const float* blin = (const float*)d_in[16];
    float* out = (float*)d_out;

    // Workspace carve-up (~90 MB total).
    char*  ws  = (char*)d_ws;
    size_t off = 0;
    auto carve = [&](size_t bytes) -> void* {
        void* p = ws + off;
        off = (off + bytes + 255) & ~(size_t)255;
        return p;
    };
    _Float16* xin16 = (_Float16*)carve((size_t)Bc * Sc * IN2c * 2); // 78.6 MB
    _Float16* h16   = (_Float16*)carve((size_t)Bc * Hc * 2);
    _Float16* hm16  = (_Float16*)carve((size_t)Bc * Hc * 2);
    _Float16* xt16  = (_Float16*)carve((size_t)Bc * IN2c * 2);
    float*    c32   = (float*)carve((size_t)Bc * Hc * 4);
    float*    gates = (float*)carve((size_t)Bc * 4 * Hc * 4);
    float*    hring = (float*)carve((size_t)5 * Bc * Hc * 4);
    float*    cmax  = (float*)carve((size_t)Bc * 3 * FNc * 4);

    // 0) init state
    init_state<<<(Bc * Hc + 255) / 256, 256, 0, stream>>>(h16, c32, cmax, Bc * Hc, Bc * 3 * FNc);

    // 1) input projection: xin = tanh(x @ W_in + b_in), stored f16
    {
        EpiArgs ea = {};
        ea.b1 = b_in; ea.out16 = xin16; ea.out16_ld = IN2c;
        dim3 g((IN2c + BN - 1) / BN, (Bc * Sc + BM - 1) / BM);
        gemm_wmma<float, float, EPI_TANH16><<<g, NTHREADS, 0, stream>>>(
            x, INc, W_in, IN2c, Bc * Sc, IN2c, INc, ea);
    }

    // 2) recurrence
    for (int t = 0; t < Sc; ++t) {
        { // xt = 2*sigmoid(h @ Q) * xin[:, t, :]
            EpiArgs ea = {};
            ea.aux = xin16 + (size_t)t * IN2c; ea.aux_ld = Sc * IN2c;
            ea.out16 = xt16; ea.out16_ld = IN2c;
            dim3 g((IN2c + BN - 1) / BN, (Bc + BM - 1) / BM);
            gemm_wmma<_Float16, float, EPI_MOGX><<<g, NTHREADS, 0, stream>>>(
                h16, Hc, Q, IN2c, Bc, IN2c, Hc, ea);
        }
        { // hm = 2*sigmoid(xt @ R) * h
            EpiArgs ea = {};
            ea.aux = h16; ea.aux_ld = Hc; ea.out16 = hm16; ea.out16_ld = Hc;
            dim3 g((Hc + BN - 1) / BN, (Bc + BM - 1) / BM);
            gemm_wmma<_Float16, float, EPI_MOGH><<<g, NTHREADS, 0, stream>>>(
                xt16, IN2c, R, Hc, Bc, Hc, IN2c, ea);
        }
        { // gates = xt @ Wih + (bih + bhh)
            EpiArgs ea = {};
            ea.b1 = bih; ea.b2 = bhh; ea.outf = gates; ea.outf_ld = 4 * Hc;
            dim3 g((4 * Hc + BN - 1) / BN, (Bc + BM - 1) / BM);
            gemm_wmma<_Float16, float, EPI_GATES1><<<g, NTHREADS, 0, stream>>>(
                xt16, IN2c, Wih, 4 * Hc, Bc, 4 * Hc, IN2c, ea);
        }
        { // gates += hm @ Whh
            EpiArgs ea = {};
            ea.outf = gates; ea.outf_ld = 4 * Hc;
            dim3 g((4 * Hc + BN - 1) / BN, (Bc + BM - 1) / BM);
            gemm_wmma<_Float16, float, EPI_GATES2><<<g, NTHREADS, 0, stream>>>(
                hm16, Hc, Whh, 4 * Hc, Bc, 4 * Hc, Hc, ea);
        }
        lstm_cell<<<(Bc * Hc + 255) / 256, 256, 0, stream>>>(gates, c32, h16, hring, t);
        if (t >= 2)
            conv_update<<<Bc, 256, 0, stream>>>(hring, Wc3, bc3, Wc4, bc4, Wc5, bc5, cmax, t);
    }

    // 3) classifier head
    final_linear<<<(Bc * 2 + 255) / 256, 256, 0, stream>>>(cmax, Wlin, blin, out);
}
